// KMeans_Torch_27900107554845
// MI455X (gfx1250) — matrix operand order repaired and checked by reading, not compiled, not run on hardware
//
#include <hip/hip_runtime.h>
#include <hip/hip_bf16.h>

typedef __attribute__((ext_vector_type(16))) _Float16 v16h;
typedef __attribute__((ext_vector_type(8)))  float    v8f;

#define D 256
#define NCLUST 2
#define MAX_ITER 100
#define BLOCK 256
#define WAVES_PER_BLOCK 8

// Workspace layout (as float indices into d_ws):
//   [0,512)     centers  (2 x 256)
//   [512,1024)  sums     (2 x 256)
//   [1024,1026) counts   (2)
//   [1026]      bias
//   [1028,1156) wh = fp16 copy of w = c1 - c0 (256 halves = 128 floats)

// ---- init: centers = rows 0 and 1 of x (JAX permutation not reproducible on-device)
__global__ __launch_bounds__(BLOCK) void k_init(const float* __restrict__ x,
                                                float* __restrict__ ws) {
    int d = threadIdx.x;
    ws[d]       = x[0 * D + d];
    ws[D + d]   = x[1 * D + d];
}

// ---- per-iteration prepare: w, wh (fp16), bias, zero sums/counts
__global__ __launch_bounds__(BLOCK) void k_prepare(float* __restrict__ ws) {
    __shared__ float red[BLOCK];
    float*    centers = ws;
    float*    sums    = ws + 512;
    float*    counts  = ws + 1024;
    float*    bias    = ws + 1026;
    _Float16* wh      = (_Float16*)(ws + 1028);

    int d = threadIdx.x;
    float c0 = centers[d];
    float c1 = centers[D + d];
    wh[d]  = (_Float16)(c1 - c0);
    red[d] = c1 * c1 - c0 * c0;
    sums[d]     = 0.0f;
    sums[D + d] = 0.0f;
    if (d < NCLUST) counts[d] = 0.0f;
    __syncthreads();
    if (d == 0) {
        float s = 0.0f;
        for (int i = 0; i < D; ++i) s += red[i];
        bias[0] = 0.5f * s;   // assign cluster 1 iff x.w > bias
    }
}

// ---- main streaming kernel: WMMA projection + segmented accumulation
__global__ __launch_bounds__(BLOCK) void k_assign(const float* __restrict__ x,
                                                  float* __restrict__ ws,
                                                  int n, int numTiles) {
    float*           sums   = ws + 512;
    float*           counts = ws + 1024;
    const _Float16*  wh     = (const _Float16*)(ws + 1028);

    __shared__ float    s_acc[WAVES_PER_BLOCK * NCLUST * D];   // 16 KB
    __shared__ _Float16 s_wh[D];
    __shared__ float    s_cnt[NCLUST];

    const int tid = threadIdx.x;
    for (int i = tid; i < WAVES_PER_BLOCK * NCLUST * D; i += BLOCK) s_acc[i] = 0.0f;
    if (tid < NCLUST) s_cnt[tid] = 0.0f;
    s_wh[tid] = wh[tid];           // BLOCK == D
    __syncthreads();

    const float bias_v = ws[1026];
    const int wave = tid >> 5;
    const int lane = tid & 31;
    const int half = lane >> 4;          // lane group: 0 => rows M 0..7, 1 => rows 8..15 in C
    const int mrow = lane & 15;          // this lane's A-matrix row

    const int globalWave = blockIdx.x * WAVES_PER_BLOCK + wave;
    const int waveStride = gridDim.x * WAVES_PER_BLOCK;

    float cnt0 = 0.0f, cnt1 = 0.0f;
    float* acc = &s_acc[wave * NCLUST * D];

    for (int t = globalWave; t < numTiles; t += waveStride) {
        const int base  = t * 16;
        int valid = n - base;  if (valid > 16) valid = 16;

        // ---------- Phase A: C[M][*] = x_M . w via 8x v_wmma_f32_16x16x32_f16 ----------
        int rA = base + mrow;  if (rA >= n) rA = n - 1;
        const float* xrow = x + (size_t)rA * D;
        v8f c = {};
        #pragma unroll
        for (int kc = 0; kc < 8; ++kc) {
            const int kb = kc * 32 + (half ? 8 : 0);
            v16h a, b;
            #pragma unroll
            for (int e = 0; e < 8; ++e) {
                a[e]     = (_Float16)xrow[kb + e];        // K = kb .. kb+7
                a[e + 8] = (_Float16)xrow[kb + 16 + e];   // K = kb+16 .. kb+23
            }
            const _Float16* wp = &s_wh[kc * 32 + (half ? 8 : 0)];
            #pragma unroll
            for (int e = 0; e < 8; ++e) { b[e] = wp[e]; b[e + 8] = wp[16 + e]; }    // all 16 B columns = w slice
            c = __builtin_amdgcn_wmma_f32_16x16x32_f16(
                    false, a, false, b, (short)0, c, false, false);
        }

        // per-lane mask for its 8 rows (r + 8*half); wave-uniform within lane groups
        unsigned m8 = 0;
        #pragma unroll
        for (int r = 0; r < 8; ++r)
            if (c[r] > bias_v) m8 |= (1u << r);
        unsigned lo = (unsigned)__shfl((int)m8, 0, 32);    // rows 0..7
        unsigned hi = (unsigned)__shfl((int)m8, 16, 32);   // rows 8..15
        unsigned mask16 = (lo & 0xffu) | ((hi & 0xffu) << 8);

        unsigned vmask = (valid >= 16) ? 0xffffu : ((1u << valid) - 1u);
        int p1 = __popc(mask16 & vmask);
        cnt1 += (float)p1;
        cnt0 += (float)(valid - p1);

        // ---------- Phase B: accumulate rows into wave-private LDS (L2-hot reload) ----------
        for (int mi = 0; mi < valid; ++mi) {
            const int cl = (mask16 >> mi) & 1;
            const float* row = x + (size_t)(base + mi) * D;
            float* a2 = acc + cl * D;
            #pragma unroll
            for (int j = 0; j < 8; ++j) {
                const int d = lane + 32 * j;               // coalesced 128B per load
                a2[d] += row[d];                           // lane-private LDS slot
            }
        }
    }

    if (lane == 0) {
        atomicAdd(&s_cnt[0], cnt0);
        atomicAdd(&s_cnt[1], cnt1);
    }
    __syncthreads();

    // block-level flush: 512 global float atomics per block
    for (int i = tid; i < NCLUST * D; i += BLOCK) {
        float s = 0.0f;
        #pragma unroll
        for (int w = 0; w < WAVES_PER_BLOCK; ++w) s += s_acc[w * NCLUST * D + i];
        atomicAdd(&sums[i], s);
    }
    if (tid == 0) {
        atomicAdd(&counts[0], s_cnt[0]);
        atomicAdd(&counts[1], s_cnt[1]);
    }
}

// ---- center update (keep old center when a cluster is empty)
__global__ __launch_bounds__(BLOCK) void k_update(float* __restrict__ ws) {
    int d = threadIdx.x;
    float* centers = ws;
    float* sums    = ws + 512;
    float* counts  = ws + 1024;
    float c0 = counts[0], c1 = counts[1];
    if (c0 > 0.0f) centers[d]     = sums[d]     / c0;
    if (c1 > 0.0f) centers[D + d] = sums[D + d] / c1;
}

// ---- final: majority-cluster mean straight from last iteration's sums/counts
__global__ __launch_bounds__(BLOCK) void k_final(const float* __restrict__ ws,
                                                 float* __restrict__ out, int n) {
    int d = threadIdx.x;
    const float* sums   = ws + 512;
    const float* counts = ws + 1024;
    int maj = (counts[1] > 0.5f * (float)n) ? 1 : 0;   // mean(assign) > 0.5
    float c = counts[maj];
    if (c < 1.0f) c = 1.0f;
    out[d] = sums[maj * D + d] / c;
}

extern "C" void kernel_launch(void* const* d_in, const int* in_sizes, int n_in,
                              void* d_out, int out_size, void* d_ws, size_t ws_size,
                              hipStream_t stream) {
    const float* x  = (const float*)d_in[0];
    float* out      = (float*)d_out;
    float* ws       = (float*)d_ws;

    const int n = in_sizes[0] / D;                 // 200000
    const int numTiles = (n + 15) / 16;            // 12500
    int blocks = (numTiles + WAVES_PER_BLOCK - 1) / WAVES_PER_BLOCK;  // 1563
    if (blocks > 4096) blocks = 4096;              // grid-stride safety

    k_init<<<1, BLOCK, 0, stream>>>(x, ws);
    for (int it = 0; it < MAX_ITER; ++it) {
        k_prepare<<<1, BLOCK, 0, stream>>>(ws);
        k_assign<<<blocks, BLOCK, 0, stream>>>(x, ws, n, numTiles);
        k_update<<<1, BLOCK, 0, stream>>>(ws);
    }
    k_final<<<1, BLOCK, 0, stream>>>(ws, out, n);
}
